// WindowAttention_21053929685586
// MI455X (gfx1250) — matrix operand order repaired and checked by reading, not compiled, not run on hardware
//
#include <hip/hip_runtime.h>
#include <hip/hip_bf16.h>

typedef __attribute__((ext_vector_type(16))) __bf16 v16bf;
typedef __attribute__((ext_vector_type(8)))  __bf16 v8bf;
typedef __attribute__((ext_vector_type(4)))  __bf16 v4bf;
typedef __attribute__((ext_vector_type(8)))  float  v8f;

// Native hardware convert (v_cvt_pk_bf16_f32 confirmed on gfx1250).
__device__ __forceinline__ __bf16 f2bf(float f) { return (__bf16)f; }

// A-operand loader: contiguous K order (matches B loads):
// lane-group g (lane>>4) holds K = {g*16 .. g*16+15}.
__device__ __forceinline__ v16bf load_a(const __bf16* rowk, int lg) {
  union { v16bf v; v8bf h[2]; } u;
  u.h[0] = *(const v8bf*)(rowk + lg * 16);
  u.h[1] = *(const v8bf*)(rowk + lg * 16 + 8);
  return u.v;
}

// ---------------- weight pre-convert (fp32 -> bf16, once per launch) ----------------
__global__ void wa_convert(const float* __restrict__ wq,
                           const float* __restrict__ wkv,
                           const float* __restrict__ wp,
                           __bf16* __restrict__ wqkv_b,   // [768][256]
                           __bf16* __restrict__ wp_b) {   // [256][256]
  int i = blockIdx.x * 256 + threadIdx.x;                 // 0 .. 262143
  if (i < 196608) {
    int row = i >> 8;
    float v = (row < 256) ? wq[i] : wkv[i - 65536];
    wqkv_b[i] = f2bf(v);
  } else {
    int j = i - 196608;                                   // 0 .. 65535
    wp_b[j] = f2bf(wp[j]);
  }
}

// ---------------- fused window attention: 1 block = 1 window ----------------
// LDS map (dynamic, 138304 B):
//   [     0,  32768) xs  [64][256] bf16      (aliased later by ps)
//   [ 32768,  65536) qs  [8][64][32] bf16    (aliased later by ps)
//   [ 65536,  98304) ks  [8][64][32] bf16    (aliased later by os)
//   [ 98304, 131072) vT  [8][32][64] bf16
//   [131072, 138272) rpb [225*8] f32
__global__ __launch_bounds__(256) void wa_fused(
    const float* __restrict__ x,        // [B][64][256]
    const __bf16* __restrict__ wqkv,    // [768][256] bf16 (rows: 256 q, 512 kv)
    const __bf16* __restrict__ wpb,     // [256][256] bf16
    const float* __restrict__ bq,       // [256]
    const float* __restrict__ bkv,      // [512]
    const float* __restrict__ bp,       // [256]
    const float* __restrict__ rpb,      // [225*8]
    float* __restrict__ out)            // [B][64][256]
{
  extern __shared__ char smem[];
  __bf16* xs   = (__bf16*)(smem);
  __bf16* qs   = (__bf16*)(smem + 32768);
  __bf16* ks   = (__bf16*)(smem + 65536);
  __bf16* vT   = (__bf16*)(smem + 98304);
  float*  rpbs = (float* )(smem + 131072);
  __bf16* ps   = (__bf16*)(smem);            // alias xs+qs : [8][64][64]
  __bf16* os   = (__bf16*)(smem + 65536);    // alias ks    : [64][256]

  const int tid  = threadIdx.x;
  // Force the wave id into an SGPR: it IS wave-uniform, but the compiler
  // can't prove it. This turns all nT/h tests into scalar branches and moves
  // tile base addressing to the SALU.
  const int wave = __builtin_amdgcn_readfirstlane(tid >> 5);
  const int lane = tid & 31;
  const int lr   = lane & 15;   // row (A) / col (B,D) within a 16-wide tile
  const int lg   = lane >> 4;   // lane half
  const int b    = blockIdx.x;

  // ---- prologue: x -> bf16 LDS (row-major), rpb table -> LDS ----
  {
    const float4* x4 = (const float4*)(x + (size_t)b * 16384);
    #pragma unroll
    for (int j = 0; j < 16; ++j) {
      int vi = j * 256 + tid;                // 0..4095 float4s
      float4 v = x4[vi];
      v4bf p;
      p[0] = f2bf(v.x); p[1] = f2bf(v.y); p[2] = f2bf(v.z); p[3] = f2bf(v.w);
      *(v4bf*)(xs + vi * 4) = p;             // one 8B DS store
    }
    for (int i = tid; i < 1800; i += 256) rpbs[i] = rpb[i];
  }
  __syncthreads();

  // ---- phase 1: fused QKV projection, Y[64][768] = Xbf16 @ Wqkv^T + b ----
  // mi-outer: cache the 8 A fragments of this row-block in registers and
  // sweep the wave's 6 N-tiles against them. Loops kept rolled to bound
  // VGPR pressure (<256, no VGPR-MSB path).
  #pragma unroll 1
  for (int mi = 0; mi < 4; ++mi) {
    v16bf A[8];
    #pragma unroll
    for (int kk = 0; kk < 8; ++kk)
      A[kk] = load_a(xs + (mi * 16 + lr) * 256 + kk * 32, lg);
    const int m0 = mi * 16 + lg * 8;         // D tile: lanes 16-31 hold M+8

    #pragma unroll 1
    for (int tt = 0; tt < 6; ++tt) {
      const int nT = wave * 6 + tt;          // scalar tile column
      const int n  = nT * 16 + lr;           // output channel 0..767
      const float bias = (nT < 16) ? bq[n] : bkv[n - 256];
      v8f acc;
      #pragma unroll
      for (int r = 0; r < 8; ++r) acc[r] = bias;
      const __bf16* brow = wqkv + (size_t)n * 256;
      #pragma unroll
      for (int kk = 0; kk < 8; ++kk) {
        v16bf bm = *(const v16bf*)(brow + kk * 32 + lg * 16);
        acc = __builtin_amdgcn_wmma_f32_16x16x32_bf16(false, A[kk], false, bm,
                                                      (short)0, acc, false, false);
      }
      if (nT < 16) {            // Q (pre-scaled by d^-0.5): scalar b16 stores
        const int h = n >> 5, d = n & 31;
        #pragma unroll
        for (int r = 0; r < 8; ++r)
          qs[(h * 64 + m0 + r) * 32 + d] = f2bf(acc[r] * 0.17677669529663689f);
      } else if (nT < 32) {     // K, [head][token][d] (B-operand of QK^T)
        const int h = (n - 256) >> 5, d = (n - 256) & 31;
        #pragma unroll
        for (int r = 0; r < 8; ++r)
          ks[(h * 64 + m0 + r) * 32 + d] = f2bf(acc[r]);
      } else {                  // V transposed [head][d][token]: one b128 store
        const int h = (n - 512) >> 5, d = (n - 512) & 31;
        v8bf pk;
        #pragma unroll
        for (int r = 0; r < 8; ++r) pk[r] = f2bf(acc[r]);
        *(v8bf*)(vT + (h * 32 + d) * 64 + m0) = pk;
      }
    }
  }
  __syncthreads();

  // ---- phase 2: per-head attention (wave = head), S = q*scale @ k^T ----
  const int h = wave;                        // scalar head id
  v8f S[4][4];
  #pragma unroll
  for (int mi = 0; mi < 4; ++mi) {
    v16bf a = load_a(qs + (h * 64 + mi * 16 + lr) * 32, lg);
    #pragma unroll
    for (int ni = 0; ni < 4; ++ni) {
      v16bf bm = *(const v16bf*)(ks + (h * 64 + ni * 16 + lr) * 32 + lg * 16);
      v8f c = {0.f, 0.f, 0.f, 0.f, 0.f, 0.f, 0.f, 0.f};
      S[mi][ni] = __builtin_amdgcn_wmma_f32_16x16x32_bf16(false, a, false, bm,
                                                          (short)0, c, false, false);
    }
  }

  // relative position bias (index recomputed arithmetically) + row softmax.
  // Row i lives in one 16-lane half; reduce with xor masks 1,2,4,8 only.
  #pragma unroll
  for (int mi = 0; mi < 4; ++mi) {
    #pragma unroll
    for (int r = 0; r < 8; ++r) {
      const int i  = mi * 16 + lg * 8 + r;
      const int iy = i >> 3, ix = i & 7;
      float mx = -3.0e38f;
      #pragma unroll
      for (int ni = 0; ni < 4; ++ni) {
        const int j   = ni * 16 + lr;
        const int rel = (iy - (j >> 3) + 7) * 15 + (ix - (j & 7) + 7);
        S[mi][ni][r] += rpbs[rel * 8 + h];
        mx = fmaxf(mx, S[mi][ni][r]);
      }
      #pragma unroll
      for (int msk = 1; msk < 16; msk <<= 1)
        mx = fmaxf(mx, __shfl_xor(mx, msk, 32));
      float sum = 0.f;
      #pragma unroll
      for (int ni = 0; ni < 4; ++ni) {
        float e = __expf(S[mi][ni][r] - mx);
        S[mi][ni][r] = e;
        sum += e;
      }
      #pragma unroll
      for (int msk = 1; msk < 16; msk <<= 1)
        sum += __shfl_xor(sum, msk, 32);
      const float inv = 1.0f / sum;
      #pragma unroll
      for (int ni = 0; ni < 4; ++ni) S[mi][ni][r] *= inv;
    }
  }
  __syncthreads();   // all waves done reading qs/xs before P overwrites them

  #pragma unroll
  for (int mi = 0; mi < 4; ++mi)
    #pragma unroll
    for (int ni = 0; ni < 4; ++ni)
      #pragma unroll
      for (int r = 0; r < 8; ++r)
        ps[(h * 64 + mi * 16 + lg * 8 + r) * 64 + ni * 16 + lr] = f2bf(S[mi][ni][r]);
  // (each wave only re-reads its own ps[h]; same-wave DS ops are ordered)

  // ---- phase 3: O = P @ V  ([64,64] x [64,32], K split into 2x32) ----
  #pragma unroll
  for (int mi = 0; mi < 4; ++mi) {
    v16bf a0 = load_a(ps + (h * 64 + mi * 16 + lr) * 64, lg);
    v16bf a1 = load_a(ps + (h * 64 + mi * 16 + lr) * 64 + 32, lg);
    #pragma unroll
    for (int ni = 0; ni < 2; ++ni) {
      v8f c = {0.f, 0.f, 0.f, 0.f, 0.f, 0.f, 0.f, 0.f};
      v16bf b0 = *(const v16bf*)(vT + (h * 32 + ni * 16 + lr) * 64 + lg * 16);
      v16bf b1 = *(const v16bf*)(vT + (h * 32 + ni * 16 + lr) * 64 + 32 + lg * 16);
      c = __builtin_amdgcn_wmma_f32_16x16x32_bf16(false, a0, false, b0, (short)0, c, false, false);
      c = __builtin_amdgcn_wmma_f32_16x16x32_bf16(false, a1, false, b1, (short)0, c, false, false);
      const int col = h * 32 + ni * 16 + lr;
      #pragma unroll
      for (int r = 0; r < 8; ++r)
        os[(mi * 16 + lg * 8 + r) * 256 + col] = f2bf(c[r]);
    }
  }
  __syncthreads();

  // ---- phase 4: output projection Z[64][256] = O @ wp^T + bp -> global ----
  #pragma unroll 1
  for (int mi = 0; mi < 4; ++mi) {
    v16bf A[8];
    #pragma unroll
    for (int kk = 0; kk < 8; ++kk)
      A[kk] = load_a(os + (mi * 16 + lr) * 256 + kk * 32, lg);

    #pragma unroll 1
    for (int tt = 0; tt < 2; ++tt) {
      const int nT = wave * 2 + tt;
      const int n  = nT * 16 + lr;
      const float bb = bp[n];
      v8f acc;
      #pragma unroll
      for (int r = 0; r < 8; ++r) acc[r] = bb;
      const __bf16* brow = wpb + (size_t)n * 256;
      #pragma unroll
      for (int kk = 0; kk < 8; ++kk) {
        v16bf bm = *(const v16bf*)(brow + kk * 32 + lg * 16);
        acc = __builtin_amdgcn_wmma_f32_16x16x32_bf16(false, A[kk], false, bm,
                                                      (short)0, acc, false, false);
      }
      float* op = out + ((size_t)b * 64 + mi * 16 + lg * 8) * 256 + n;
      #pragma unroll
      for (int r = 0; r < 8; ++r) op[(size_t)r * 256] = acc[r];
    }
  }
}

extern "C" void kernel_launch(void* const* d_in, const int* in_sizes, int n_in,
                              void* d_out, int out_size, void* d_ws, size_t ws_size,
                              hipStream_t stream) {
  const float* x   = (const float*)d_in[0];
  const float* wq  = (const float*)d_in[1];
  const float* bq  = (const float*)d_in[2];
  const float* wkv = (const float*)d_in[3];
  const float* bkv = (const float*)d_in[4];
  const float* wp  = (const float*)d_in[5];
  const float* bp  = (const float*)d_in[6];
  const float* rpb = (const float*)d_in[7];

  __bf16* wqkv_b = (__bf16*)d_ws;                                   // 393216 B
  __bf16* wp_b   = (__bf16*)((char*)d_ws + 768 * 256 * 2);          // 131072 B

  wa_convert<<<1024, 256, 0, stream>>>(wq, wkv, wp, wqkv_b, wp_b);

  const int ldsBytes = 138304;   // 4*32KB tiles + 7.2KB rpb table
  hipFuncSetAttribute((const void*)wa_fused,
                      hipFuncAttributeMaxDynamicSharedMemorySize, ldsBytes);
  wa_fused<<<4096, 256, ldsBytes, stream>>>(x, wqkv_b, wp_b, bq, bkv, bp, rpb,
                                            (float*)d_out);
  (void)in_sizes; (void)n_in; (void)out_size; (void)ws_size;
}
